// BlockSparseAttention_26250840113199
// MI455X (gfx1250) — compile-verified, matrix operand order checked by reading, not run on hardware
//
#include <hip/hip_runtime.h>
#include <hip/hip_bf16.h>

// =====================================================================
// BlockSparseAttention for MI455X (gfx1250, wave32, WMMA).
//
// Memory floor: the dense attn output [2,16,2048,2048] fp32 = 537 MB of
// mandatory writes -> ~23us at 23.3 TB/s. All matmuls (~70 GFLOP total)
// are negligible next to that on WMMA bf16, so the design streams attn
// exactly once with non-temporal stores and skips WMMA work for masked
// 32-column chunks (masked probs underflow to exact 0.0f, matching the
// fp32 reference: expf(-10000-m) == 0).
//
// CDNA5-specific paths used:
//   - v_wmma_f32_16x16x32_bf16 for all four matmuls
//   - global_load_async_to_lds_b128 (+ s_wait_asynccnt) for LDS staging
//   - non-temporal (TH=NT) stores for the 537 MB attn stream
//   - ds_load_b128 fragment gathers matching the ISA 16-bit A/B layouts
//
// Workspace layout (needs 32 MB of d_ws):
//   [ 0MB..8MB)  Q  bf16 [b,h,s,64]
//   [ 8MB..16MB) K  bf16 [b,h,s,64]
//   [16MB..24MB) V^T bf16 [b,h,64,s]   (pre-transposed -> WMMA B layout)
//   [24MB..32MB) O  bf16 [b,s,1024]    (heads concatenated)
// =====================================================================

#define D_MODEL 1024
#define N_HEADS 16
#define HEAD_DIM 64
#define SEQ 2048
#define BATCH 2
#define HALF_WIN 64
#define SCALE 0.125f

typedef __attribute__((ext_vector_type(16))) __bf16 v16bf;
typedef __attribute__((ext_vector_type(8)))  float  v8f;

static __device__ inline v8f zero8() {
    v8f z = {0.f,0.f,0.f,0.f,0.f,0.f,0.f,0.f};
    return z;
}

// Load one 16-lane WMMA operand (A layout, or B via row-major B^T) from a
// row-major bf16 tile. Caller passes the per-lane pointer:
//   p = tile + (lane&15)*stride + (lane>>4)*8
// elements 0..7  <- p[0..7]   (k = ksel*8 .. +7)
// elements 8..15 <- p[16..23] (k = 16 + ksel*8 .. +7)
// This matches the CDNA5 16-bit A-matrix 16x32 VGPR layout (ISA 7.12.2).
static __device__ inline v16bf load_frag16x32(const __bf16* p) {
    v16bf f;
    uint4* fp = reinterpret_cast<uint4*>(&f);
    fp[0] = *reinterpret_cast<const uint4*>(p);
    fp[1] = *reinterpret_cast<const uint4*>(p + 16);
    return f;
}

static __device__ inline v8f wmma_bf16(v16bf a, v16bf b, v8f c) {
    return __builtin_amdgcn_wmma_f32_16x16x32_bf16(
        /*neg_a=*/false, a, /*neg_b=*/false, b,
        /*c_mod=*/(short)0, c, /*reuse_a=*/false, /*reuse_b=*/false);
}

// Async DMA of 16 bytes/lane from global memory into LDS (CDNA5 Tensor/
// async path, tracked by ASYNCcnt). lds_off is the per-lane LDS byte
// address (low 32 bits of a generic pointer to __shared__).
static __device__ inline void async_copy_b128(unsigned lds_off, const void* gaddr) {
    asm volatile("global_load_async_to_lds_b128 %0, %1, off"
                 :: "v"(lds_off), "v"(gaddr) : "memory");
}
static __device__ inline void wait_asynccnt0() {
    asm volatile("s_wait_asynccnt 0" ::: "memory");
}
static __device__ inline unsigned lds_addr(const void* p) {
    return (unsigned)(unsigned long long)p;   // low 32 bits = LDS byte offset
}

// ---------------------------------------------------------------------
// Kernel 1: fused QKV projection.  C[m][n] = sum_k x[m][k]*W[n][k] + b[n]
// M=4096, N=3072, K=1024.  Block tile 128x64, 8 waves, each wave 64x16.
// Epilogue routes columns to Q/K (row-major [bh,s,d]) or V (transposed
// [bh,d,s]) as bf16.  (fp32 inputs need VALU conversion, so staging goes
// through VGPRs here.)
// ---------------------------------------------------------------------
__global__ __launch_bounds__(256) void qkv_gemm_kernel(
    const float* __restrict__ x, const float* __restrict__ w,
    const float* __restrict__ bias,
    __bf16* __restrict__ qbuf, __bf16* __restrict__ kbuf,
    __bf16* __restrict__ vT)
{
    __shared__ __bf16 sA[128 * 32];
    __shared__ __bf16 sB[64 * 32];
    const int tid  = threadIdx.x;
    const int lane = tid & 31, wave = tid >> 5;
    const int wm = wave >> 2, wn = wave & 3;   // 2 x 4 wave grid
    const int lr = lane & 15, ks = lane >> 4;
    const int m0 = blockIdx.x * 128, n0 = blockIdx.y * 64;

    v8f acc[4];
#pragma unroll
    for (int t = 0; t < 4; ++t) acc[t] = zero8();

    for (int k0 = 0; k0 < D_MODEL; k0 += 32) {
        // stage A: 128x32 fp32 -> bf16 (1024 float4, 4 per thread)
#pragma unroll
        for (int i = 0; i < 4; ++i) {
            int e = tid + i * 256;
            int row = e >> 3, c4 = e & 7;
            float4 v = *reinterpret_cast<const float4*>(
                x + (size_t)(m0 + row) * D_MODEL + k0 + c4 * 4);
            __bf16* d = &sA[row * 32 + c4 * 4];
            d[0] = (__bf16)v.x; d[1] = (__bf16)v.y;
            d[2] = (__bf16)v.z; d[3] = (__bf16)v.w;
        }
        // stage B (W rows = output features): 64x32 fp32 -> bf16
#pragma unroll
        for (int i = 0; i < 2; ++i) {
            int e = tid + i * 256;
            int row = e >> 3, c4 = e & 7;
            float4 v = *reinterpret_cast<const float4*>(
                w + (size_t)(n0 + row) * D_MODEL + k0 + c4 * 4);
            __bf16* d = &sB[row * 32 + c4 * 4];
            d[0] = (__bf16)v.x; d[1] = (__bf16)v.y;
            d[2] = (__bf16)v.z; d[3] = (__bf16)v.w;
        }
        __syncthreads();
        v16bf bfrag = load_frag16x32(&sB[(wn * 16 + lr) * 32 + ks * 8]);
#pragma unroll
        for (int t = 0; t < 4; ++t) {
            v16bf afrag = load_frag16x32(&sA[(wm * 64 + t * 16 + lr) * 32 + ks * 8]);
            acc[t] = wmma_bf16(afrag, bfrag, acc[t]);
        }
        __syncthreads();
    }

    // epilogue: C layout elem r -> (M = r + 8*(lane>>4), N = lane&15)
    const int gn = n0 + wn * 16 + lr;
    const float bv = bias[gn];
    const int sec = gn >> 10;          // 0=Q 1=K 2=V
    const int f = gn & 1023;
    const int h = f >> 6, d = f & 63;
#pragma unroll
    for (int t = 0; t < 4; ++t) {
#pragma unroll
        for (int r = 0; r < 8; ++r) {
            int gm = m0 + wm * 64 + t * 16 + r + 8 * ks;
            int b = gm >> 11, sidx = gm & 2047;
            int bh = b * N_HEADS + h;
            __bf16 val = (__bf16)(acc[t][r] + bv);
            if (sec == 0)
                qbuf[((size_t)(bh * SEQ + sidx)) * HEAD_DIM + d] = val;
            else if (sec == 1)
                kbuf[((size_t)(bh * SEQ + sidx)) * HEAD_DIM + d] = val;
            else
                vT[((size_t)(bh * HEAD_DIM + d)) * SEQ + sidx] = val;
        }
    }
}

// ---------------------------------------------------------------------
// Kernel 2: sparse attention. One block = (bh, 16 query rows).
// 8 waves each own chunks of 32 key columns strided by 8.
// Two passes: (1) online row max/sumexp over active chunks,
// (2) recompute scores, NT-store probabilities, accumulate O via WMMA.
// ---------------------------------------------------------------------
__global__ __launch_bounds__(256) void attn_kernel(
    const __bf16* __restrict__ qbuf, const __bf16* __restrict__ kbuf,
    const __bf16* __restrict__ vT, const unsigned char* __restrict__ lm,
    float* __restrict__ attn, __bf16* __restrict__ obuf)
{
    __shared__ __bf16 sQ[16 * 64];
    __shared__ unsigned char sAct[64];   // per-32-col-chunk landmark activity
    __shared__ float  sM[8][16];
    __shared__ float  sL[8][16];
    __shared__ float  sFM[16];
    __shared__ float  sFL[16];
    __shared__ __bf16 sP[8][16 * 32];
    __shared__ float  sOw[8][16 * 64];   // per-wave partial O (deterministic reduce)

    const int tid  = threadIdx.x;
    const int lane = tid & 31, wave = tid >> 5;
    const int lr = lane & 15, ks = lane >> 4;
    const int q0 = blockIdx.x * 16;
    const int bh = blockIdx.y;
    const int b  = bh >> 4;

    const __bf16* qb = qbuf + (size_t)bh * SEQ * HEAD_DIM;
    const __bf16* kb = kbuf + (size_t)bh * SEQ * HEAD_DIM;
    const __bf16* vb = vT   + (size_t)bh * HEAD_DIM * SEQ;
    const unsigned char* lmb = lm + b * SEQ;   // landmark bool mask [b,1,s]
    float* attnB = attn + (size_t)bh * SEQ * SEQ;

    // stage Q tile (16x64 bf16 = 128 lanes x 16B) via async DMA to LDS
    if (tid < 128) {
        async_copy_b128(lds_addr(&sQ[tid * 8]),
                        qb + (size_t)q0 * HEAD_DIM + tid * 8);
        wait_asynccnt0();
    }
    // per-chunk landmark activity flags (computed once, used in both passes)
    if (tid < 64) {
        const uint* lw = reinterpret_cast<const uint*>(lmb + tid * 32);
        uint any = 0;
#pragma unroll
        for (int i = 0; i < 8; ++i) any |= lw[i];
        sAct[tid] = any ? 1 : 0;
    }
    __syncthreads();

    const v16bf a_lo = load_frag16x32(&sQ[lr * 64 + ks * 8]);        // d 0..31
    const v16bf a_hi = load_frag16x32(&sQ[lr * 64 + 32 + ks * 8]);   // d 32..63

    float m_r[8], l_r[8];
#pragma unroll
    for (int r = 0; r < 8; ++r) { m_r[r] = -3.0e38f; l_r[r] = 0.f; }

    // ---------------- pass 1: row statistics ----------------
    for (int c = wave; c < SEQ / 32; c += 8) {
        const int col0 = c * 32;
        bool localAct = (col0 + 31 >= q0 - HALF_WIN) && (col0 <= q0 + 15 + HALF_WIN);
        if (!(localAct || sAct[c])) continue;

        float sv[2][8];
        unsigned char lmv[2];
#pragma unroll
        for (int sub = 0; sub < 2; ++sub) {
            const __bf16* kp = kb + (size_t)(col0 + sub * 16 + lr) * HEAD_DIM + ks * 8;
            v16bf b0 = load_frag16x32(kp);
            v16bf b1 = load_frag16x32(kp + 32);
            v8f sc = zero8();
            sc = wmma_bf16(a_lo, b0, sc);
            sc = wmma_bf16(a_hi, b1, sc);
            lmv[sub] = lmb[col0 + sub * 16 + lr];
#pragma unroll
            for (int r = 0; r < 8; ++r) sv[sub][r] = sc[r];
        }
#pragma unroll
        for (int r = 0; r < 8; ++r) {
            int qi = q0 + r + 8 * ks;
            int d0 = qi - (col0 + lr);        d0 = d0 < 0 ? -d0 : d0;
            int d1 = qi - (col0 + 16 + lr);   d1 = d1 < 0 ? -d1 : d1;
            float v0 = (d0 <= HALF_WIN || lmv[0]) ? sv[0][r] * SCALE : -3.0e38f;
            float v1 = (d1 <= HALF_WIN || lmv[1]) ? sv[1][r] * SCALE : -3.0e38f;
            float vmax = fmaxf(v0, v1);
#pragma unroll
            for (int o = 8; o >= 1; o >>= 1)
                vmax = fmaxf(vmax, __shfl_xor(vmax, o, 32));
            float newM = fmaxf(m_r[r], vmax);
            float e = __expf(v0 - newM) + __expf(v1 - newM);
#pragma unroll
            for (int o = 8; o >= 1; o >>= 1)
                e += __shfl_xor(e, o, 32);
            l_r[r] = l_r[r] * __expf(m_r[r] - newM) + e;
            m_r[r] = newM;
        }
    }

    // combine stats across waves (lanes 0 and 16 publish rows 0-7 / 8-15)
    if (lr == 0) {
#pragma unroll
        for (int r = 0; r < 8; ++r) {
            sM[wave][r + 8 * ks] = m_r[r];
            sL[wave][r + 8 * ks] = l_r[r];
        }
    }
    __syncthreads();
    if (tid < 16) {
        float fm = -3.0e38f;
        for (int w = 0; w < 8; ++w) fm = fmaxf(fm, sM[w][tid]);
        float fl = 0.f;
        for (int w = 0; w < 8; ++w) fl += sL[w][tid] * __expf(sM[w][tid] - fm);
        sFM[tid] = fm;
        sFL[tid] = (fl > 0.f) ? 1.0f / fl : 0.f;
    }
    __syncthreads();
    float fm_r[8], il_r[8];
#pragma unroll
    for (int r = 0; r < 8; ++r) {
        fm_r[r] = sFM[r + 8 * ks];
        il_r[r] = sFL[r + 8 * ks];
    }

    v8f o_acc[4];
#pragma unroll
    for (int t = 0; t < 4; ++t) o_acc[t] = zero8();

    // ---------------- pass 2: probabilities + P@V ----------------
    for (int c = wave; c < SEQ / 32; c += 8) {
        const int col0 = c * 32;
        bool localAct = (col0 + 31 >= q0 - HALF_WIN) && (col0 <= q0 + 15 + HALF_WIN);
        if (!(localAct || sAct[c])) {
            // masked chunk: attn is exactly zero; coalesced NT zero-fill
#pragma unroll
            for (int row = 0; row < 16; ++row)
                __builtin_nontemporal_store(
                    0.0f, attnB + (size_t)(q0 + row) * SEQ + col0 + lane);
            continue;
        }
        __bf16* pw = &sP[wave][0];
#pragma unroll
        for (int sub = 0; sub < 2; ++sub) {
            const __bf16* kp = kb + (size_t)(col0 + sub * 16 + lr) * HEAD_DIM + ks * 8;
            v16bf b0 = load_frag16x32(kp);
            v16bf b1 = load_frag16x32(kp + 32);
            v8f sc = zero8();
            sc = wmma_bf16(a_lo, b0, sc);
            sc = wmma_bf16(a_hi, b1, sc);
            unsigned char lmv = lmb[col0 + sub * 16 + lr];
#pragma unroll
            for (int r = 0; r < 8; ++r) {
                int qi = q0 + r + 8 * ks;
                int kj = col0 + sub * 16 + lr;
                int dd = qi - kj; dd = dd < 0 ? -dd : dd;
                float val = (dd <= HALF_WIN || lmv) ? sc[r] * SCALE : -3.0e38f;
                float p = __expf(val - fm_r[r]) * il_r[r];
                __builtin_nontemporal_store(p, attnB + (size_t)qi * SEQ + kj);
                pw[(r + 8 * ks) * 32 + sub * 16 + lr] = (__bf16)p;  // transpose via LDS
            }
        }
        // A = P (16x32) from per-wave LDS; B = V^T tiles straight from global
        v16bf pa = load_frag16x32(&pw[lr * 32 + ks * 8]);
#pragma unroll
        for (int ds = 0; ds < 4; ++ds) {
            const __bf16* vp = vb + (size_t)(ds * 16 + lr) * SEQ + col0 + ks * 8;
            v16bf bv = load_frag16x32(vp);
            o_acc[ds] = wmma_bf16(pa, bv, o_acc[ds]);
        }
    }

    // deterministic cross-wave reduce of O through LDS (no float atomics)
#pragma unroll
    for (int ds = 0; ds < 4; ++ds)
#pragma unroll
        for (int r = 0; r < 8; ++r)
            sOw[wave][(r + 8 * ks) * 64 + ds * 16 + lr] = o_acc[ds][r];
    __syncthreads();

    const int h = bh & (N_HEADS - 1);
    for (int e = tid; e < 16 * 64; e += 256) {
        int row = e >> 6, d = e & 63;
        float sum = 0.f;
#pragma unroll
        for (int w = 0; w < 8; ++w) sum += sOw[w][e];
        obuf[((size_t)(b * SEQ + q0 + row)) * D_MODEL + h * HEAD_DIM + d] = (__bf16)sum;
    }
}

// ---------------------------------------------------------------------
// Kernel 3: output projection. out[m][n] = sum_k O[m][k]*Wo[n][k] + b[n]
// M=4096, N=1024, K=1024.  A tile is already bf16 -> stage it with the
// async-to-LDS DMA (ASYNCcnt) so it overlaps the fp32->bf16 conversion
// of the B tile running on the VALU in the same iteration.
// ---------------------------------------------------------------------
__global__ __launch_bounds__(256) void out_gemm_kernel(
    const __bf16* __restrict__ obuf, const float* __restrict__ w,
    const float* __restrict__ bias, float* __restrict__ out)
{
    __shared__ __bf16 sA[128 * 32];
    __shared__ __bf16 sB[64 * 32];
    const int tid  = threadIdx.x;
    const int lane = tid & 31, wave = tid >> 5;
    const int wm = wave >> 2, wn = wave & 3;
    const int lr = lane & 15, ks = lane >> 4;
    const int m0 = blockIdx.x * 128, n0 = blockIdx.y * 64;

    v8f acc[4];
#pragma unroll
    for (int t = 0; t < 4; ++t) acc[t] = zero8();

    for (int k0 = 0; k0 < D_MODEL; k0 += 32) {
        // stage A (bf16): 128 rows x 32 bf16 = 512 x 16B, 2 per thread,
        // issued as async DMA straight into LDS
#pragma unroll
        for (int i = 0; i < 2; ++i) {
            int e = tid + i * 256;
            int row = e >> 2, c = e & 3;
            async_copy_b128(lds_addr(&sA[row * 32 + c * 8]),
                            obuf + (size_t)(m0 + row) * D_MODEL + k0 + c * 8);
        }
        // stage B: fp32 -> bf16 on the VALU (overlaps the async A copy)
#pragma unroll
        for (int i = 0; i < 2; ++i) {
            int e = tid + i * 256;
            int row = e >> 3, c4 = e & 7;
            float4 v = *reinterpret_cast<const float4*>(
                w + (size_t)(n0 + row) * D_MODEL + k0 + c4 * 4);
            __bf16* d = &sB[row * 32 + c4 * 4];
            d[0] = (__bf16)v.x; d[1] = (__bf16)v.y;
            d[2] = (__bf16)v.z; d[3] = (__bf16)v.w;
        }
        wait_asynccnt0();
        __syncthreads();
        v16bf bfrag = load_frag16x32(&sB[(wn * 16 + lr) * 32 + ks * 8]);
#pragma unroll
        for (int t = 0; t < 4; ++t) {
            v16bf afrag = load_frag16x32(&sA[(wm * 64 + t * 16 + lr) * 32 + ks * 8]);
            acc[t] = wmma_bf16(afrag, bfrag, acc[t]);
        }
        __syncthreads();
    }

    const int gn = n0 + wn * 16 + lr;
    const float bv = bias[gn];
#pragma unroll
    for (int t = 0; t < 4; ++t) {
#pragma unroll
        for (int r = 0; r < 8; ++r) {
            int gm = m0 + wm * 64 + t * 16 + r + 8 * ks;
            out[(size_t)gm * D_MODEL + gn] = acc[t][r] + bv;
        }
    }
}

// ---------------------------------------------------------------------
extern "C" void kernel_launch(void* const* d_in, const int* in_sizes, int n_in,
                              void* d_out, int out_size, void* d_ws, size_t ws_size,
                              hipStream_t stream) {
    const float* x              = (const float*)d_in[0];
    const unsigned char* lmmask = (const unsigned char*)d_in[1];  // jnp bool_ (1B)
    const float* qkv_w          = (const float*)d_in[2];
    const float* qkv_b          = (const float*)d_in[3];
    const float* out_w          = (const float*)d_in[4];
    const float* out_b          = (const float*)d_in[5];

    float* out  = (float*)d_out;                              // [2,2048,1024]
    float* attn = out + (size_t)BATCH * SEQ * D_MODEL;        // [2,16,2048,2048]

    char* ws = (char*)d_ws;                                   // needs 32 MB
    __bf16* qbuf = reinterpret_cast<__bf16*>(ws);
    __bf16* kbuf = reinterpret_cast<__bf16*>(ws + ((size_t)8  << 20));
    __bf16* vT   = reinterpret_cast<__bf16*>(ws + ((size_t)16 << 20));
    __bf16* obuf = reinterpret_cast<__bf16*>(ws + ((size_t)24 << 20));

    qkv_gemm_kernel<<<dim3(32, 48), 256, 0, stream>>>(x, qkv_w, qkv_b,
                                                      qbuf, kbuf, vT);
    attn_kernel<<<dim3(SEQ / 16, BATCH * N_HEADS), 256, 0, stream>>>(
        qbuf, kbuf, vT, lmmask, attn, obuf);
    out_gemm_kernel<<<dim3(32, 16), 256, 0, stream>>>(obuf, out_w, out_b, out);
}
